// NeutronStarPhysicsGuidedPINN_21260088115673
// MI455X (gfx1250) — matrix operand order repaired and checked by reading, not compiled, not run on hardware
//
#include <hip/hip_runtime.h>

typedef __attribute__((ext_vector_type(16))) _Float16 v16h;
typedef __attribute__((ext_vector_type(8)))  float    v8f;

#define DEVFN __device__ __forceinline__

// tanh: gfx1250 V_TANH_F32 (confirmed present in round 1 asm).
DEVFN float fast_tanh(float x) {
#if __has_builtin(__builtin_amdgcn_tanhf)
  return __builtin_amdgcn_tanhf(x);
#elif __has_builtin(__builtin_amdgcn_tanh_f32)
  return __builtin_amdgcn_tanh_f32(x);
#else
  float e = __builtin_amdgcn_exp2f(2.8853900817779268f * x);  // 2*log2(e)*x
  return 1.0f - 2.0f * __builtin_amdgcn_rcpf(e + 1.0f);
#endif
}

DEVFN float fast_log1p(float y) {
  return 0.6931471805599453f * __builtin_amdgcn_logf(1.0f + y);
}

// lane i <-> lane i^16 half-swap. Prefer v_permlanex16_b32 (VALU, no DScnt)
// over __shfl_xor's ds_bpermute.
DEVFN float shx16(float v) {
#if __has_builtin(__builtin_amdgcn_permlanex16)
  int i = __builtin_bit_cast(int, v);
  i = __builtin_amdgcn_permlanex16(i, i, 0x76543210, 0xfedcba98, false, false);
  return __builtin_bit_cast(float, i);
#else
  return __shfl_xor(v, 16, 32);
#endif
}

// Build a 16x32 f16 A-fragment from a (row m, col k) -> float functor.
// Layout: lanes 0-15 hold M=m, K=0..15; lanes 16-31 hold K=16..31 (contiguous, matches B fills).
template <typename F>
DEVFN v16h build_a(int lane, F f) {
  const int m   = lane & 15;
  const int kb0 = (lane & 16) ? 16 : 0;
  const int kb1 = (lane & 16) ? 24 : 8;
  v16h a = {};
#pragma unroll
  for (int i = 0; i < 8; ++i) a[i]     = (_Float16)f(m, kb0 + i);
#pragma unroll
  for (int i = 0; i < 8; ++i) a[8 + i] = (_Float16)f(m, kb1 + i);
  return a;
}

__global__ void __launch_bounds__(256) pinn_wmma_kernel(
    const float* __restrict__ x,
    const float* __restrict__ vW1, const float* __restrict__ vb1,
    const float* __restrict__ vW2, const float* __restrict__ vb2,
    const float* __restrict__ vW3, const float* __restrict__ vb3,
    const float* __restrict__ cW1, const float* __restrict__ cb1,
    const float* __restrict__ cW2, const float* __restrict__ cb2,
    const float* __restrict__ cW3, const float* __restrict__ cb3,
    const float* __restrict__ oW1, const float* __restrict__ ob1,
    const float* __restrict__ oW2, const float* __restrict__ ob2,
    const float* __restrict__ oW3, const float* __restrict__ ob3,
    float* __restrict__ out, int n)
{
  const int lane = threadIdx.x & 31;
  const int wave = (blockIdx.x * blockDim.x + threadIdx.x) >> 5;
  const int nwav = (gridDim.x * blockDim.x) >> 5;
  const bool lo  = lane < 16;

  // ---- A fragments (weights), built once per wave. Bias folded via ones-rows in B.
  const v16h a1vc = build_a(lane, [&](int m, int k) -> float {
    if (k < 4) {
      if (m < 4)  return (k < 3) ? vW1[m * 3 + k]       : vb1[m];
      if (m < 10) return (k < 3) ? cW1[(m - 4) * 3 + k] : cb1[m - 4];
    }
    return 0.0f;
  });
  const v16h a1o = build_a(lane, [&](int m, int k) -> float {
    if (k < 4 && m < 8) return (k < 3) ? oW1[m * 3 + k] : ob1[m];
    return 0.0f;
  });
  const v16h a2 = build_a(lane, [&](int m, int k) -> float {
    if (m < 2)      { if (k < 4)              return vW2[m * 4 + k];
                      if (k == 15)            return vb2[m]; }
    else if (m < 5) { if (k >= 4  && k < 10)  return cW2[(m - 2) * 6 + (k - 4)];
                      if (k == 15)            return cb2[m - 2]; }
    else if (m < 9) { if (k >= 16 && k < 24)  return oW2[(m - 5) * 8 + (k - 16)];
                      if (k == 31)            return ob2[m - 5]; }
    return 0.0f;
  });
  const v16h a3 = build_a(lane, [&](int m, int k) -> float {
    if (m == 0)      { if (k < 2)             return vW3[k];
                       if (k == 15)           return vb3[0]; }
    else if (m == 1) { if (k >= 2 && k < 5)   return cW3[k - 2];
                       if (k == 15)           return cb3[0]; }
    else if (m == 2) { if (k >= 5 && k < 9)   return oW3[k - 5];
                       if (k == 15)           return ob3[0]; }
    return 0.0f;
  });

  const int step = nwav * 32;
  int base = wave * 32;
  if (base >= n) return;                     // uniform per wave

  // Software-pipelined x loads (index clamped so EXEC stays all-1 for WMMA).
  float D, q, r;
  {
    const int p = min(base + lane, n - 1);
    D = x[3 * p + 0]; q = x[3 * p + 1]; r = x[3 * p + 2];
  }

  for (; base < n; base += step) {
    // issue next iteration's loads before the WMMA dependency chain
    const int pn = min(base + step + lane, n - 1);
    const float Dn = x[3 * pn + 0];
    const float qn = x[3 * pn + 1];
    const float rn = x[3 * pn + 2];

    float vo[2], co[2], oo[2];
#pragma unroll
    for (int t = 0; t < 2; ++t) {
      // Tile t covers points base+16t .. base+16t+15; B columns come from lanes 0-15.
      const float Dt = t ? shx16(D) : D;
      const float qt = t ? shx16(q) : q;
      const float rt = t ? shx16(r) : r;

      v16h b1 = {};
      b1[0] = (_Float16)(lo ? Dt : 0.0f);
      b1[1] = (_Float16)(lo ? qt : 0.0f);
      b1[2] = (_Float16)(lo ? rt : 0.0f);
      b1[3] = (_Float16)(lo ? 1.0f : 0.0f);

      const v8f z = {};
      v8f h_vc = __builtin_amdgcn_wmma_f32_16x16x32_f16(false, a1vc, false, b1, (short)0, z, false, false);
      v8f h_o  = __builtin_amdgcn_wmma_f32_16x16x32_f16(false, a1o,  false, b1, (short)0, z, false, false);
#pragma unroll
      for (int j = 0; j < 8; ++j) { h_vc[j] = fast_tanh(h_vc[j]); h_o[j] = fast_tanh(h_o[j]); }

      // C layout -> next B: low lanes need partner rows 8,9 of h_vc; high lanes need
      // partner rows 0-7 of h_o (placed at K=16..23). All branchless.
      const float pvc0 = shx16(h_vc[0]);
      const float pvc1 = shx16(h_vc[1]);
      float po[8];
#pragma unroll
      for (int j = 0; j < 8; ++j) po[j] = shx16(h_o[j]);

      v16h b2 = {};
#pragma unroll
      for (int j = 0; j < 8; ++j) b2[j] = (_Float16)(lo ? h_vc[j] : po[j]);
      b2[8]  = (_Float16)(lo ? pvc0 : 0.0f);
      b2[9]  = (_Float16)(lo ? pvc1 : 0.0f);
      b2[15] = (_Float16)1.0f;               // K=15 (v/c bias) and K=31 (o bias)

      v8f h2 = __builtin_amdgcn_wmma_f32_16x16x32_f16(false, a2, false, b2, (short)0, z, false, false);
#pragma unroll
      for (int j = 0; j < 8; ++j) h2[j] = fast_tanh(h2[j]);

      const float p20 = shx16(h2[0]);        // h2 row 8 for low lanes
      v16h b3 = {};
#pragma unroll
      for (int j = 0; j < 8; ++j) b3[j] = (_Float16)(lo ? h2[j] : 0.0f);
      b3[8]  = (_Float16)(lo ? p20 : 0.0f);
      b3[15] = (_Float16)(lo ? 1.0f : 0.0f);

      const v8f c3 = __builtin_amdgcn_wmma_f32_16x16x32_f16(false, a3, false, b3, (short)0, z, false, false);
      vo[t] = c3[0]; co[t] = c3[1]; oo[t] = c3[2];               // rows 0/1/2, col = point
    }

    // Tile-1 results live in lanes 0-15; route to lanes 16-31 (branchless).
    const float vo1 = shx16(vo[1]);
    const float co1 = shx16(co[1]);
    const float oo1 = shx16(oo[1]);
    const float v_out = lo ? vo[0] : vo1;
    const float c_out = lo ? co[0] : co1;
    const float o_out = lo ? oo[0] : oo1;

    // ---- physics branch math (VALU/trans, co-executes with WMMA) ----
    const float zkin  = __builtin_amdgcn_sqrtf(1.0f + r * r);
    const bool  vac   = D < 1e-8f;
    const bool  crust = !vac && (D < 1e-5f);
    const bool  core  = (D >= 1e-5f) && (D < 1e-3f);
    const float z_vac   = zkin * (1.0f + 1.5f * q);
    const float z_crust = zkin * (1.0f + 2.0f * q) * (1.0f + 0.1f * fast_log1p(D * 1.0e5f));
    const float z_core  = zkin * (1.0f + 3.0f * q) * (1.0f + 0.2f * D * __builtin_amdgcn_rcpf(1.0f + D));
    const float z_nuc   = zkin * (1.0f + 5.0f * q * __builtin_amdgcn_rcpf(1.0f + q))
                               * (1.0f + 0.5f * fast_log1p(D * 1.0e3f));
    float zg = vac ? z_vac : (crust ? z_crust : (core ? z_core : z_nuc));
    zg = fminf(fmaxf(zg, 1.0f), 100.0f);
    const float corr = vac   ? 0.05f * v_out
                     : crust ? 0.10f * c_out
                     : core  ? 0.20f * o_out
                             : 0.40f * o_out;

    if (base + lane < n) out[base + lane] = zg + corr;

    D = Dn; q = qn; r = rn;
  }
}

extern "C" void kernel_launch(void* const* d_in, const int* in_sizes, int n_in,
                              void* d_out, int out_size, void* d_ws, size_t ws_size,
                              hipStream_t stream) {
  const float* xp  = (const float*)d_in[0];
  const float* vW1 = (const float*)d_in[1];
  const float* vb1 = (const float*)d_in[2];
  const float* vW2 = (const float*)d_in[3];
  const float* vb2 = (const float*)d_in[4];
  const float* vW3 = (const float*)d_in[5];
  const float* vb3 = (const float*)d_in[6];
  const float* cW1 = (const float*)d_in[7];
  const float* cb1 = (const float*)d_in[8];
  const float* cW2 = (const float*)d_in[9];
  const float* cb2 = (const float*)d_in[10];
  const float* cW3 = (const float*)d_in[11];
  const float* cb3 = (const float*)d_in[12];
  const float* oW1 = (const float*)d_in[13];
  const float* ob1 = (const float*)d_in[14];
  const float* oW2 = (const float*)d_in[15];
  const float* ob2 = (const float*)d_in[16];
  const float* oW3 = (const float*)d_in[17];
  const float* ob3 = (const float*)d_in[18];
  float* out = (float*)d_out;

  const int n = in_sizes[0] / 3;
  const int threads = 256;                       // 8 wave32s per block
  int waves_needed = (n + 31) / 32;
  int blocks = (waves_needed + 7) / 8;
  if (blocks > 1024) blocks = 1024;
  if (blocks < 1) blocks = 1;

  pinn_wmma_kernel<<<blocks, threads, 0, stream>>>(
      xp, vW1, vb1, vW2, vb2, vW3, vb3,
      cW1, cb1, cW2, cb2, cW3, cb3,
      oW1, ob1, oW2, ob2, oW3, ob3,
      out, n);
}
